// FracBNNResidualBlock_57062935495237
// MI455X (gfx1250) — compile-verified, matrix operand order checked by reading, not run on hardware
//
#include <hip/hip_runtime.h>
#include <stdint.h>

typedef int   v16i __attribute__((ext_vector_type(16)));
typedef int   v4i  __attribute__((ext_vector_type(4)));
typedef float v8f  __attribute__((ext_vector_type(8)));

#define CC     256
#define HH     56
#define WWID   56
#define HP     58
#define KTOT   2304
#define KSTEPS 18
#define EPSV   1e-5f
#define BROW   144                    // padded LDS row stride for B tile (bank spread)
#define BBUF   (64 * BROW)            // 9216 B per buffer

__device__ __forceinline__ uint8_t sign_fp8(float x) {
  // FP8 E4M3: +1.0 = 0x38, -1.0 = 0xB8, 0.0 = 0x00
  return x > 0.f ? (uint8_t)0x38 : (x < 0.f ? (uint8_t)0xB8 : (uint8_t)0x00);
}

// ---- binarize weights: w[co][ci][3][3] f32 -> bout[co][k] fp8, k = (kh*3+kw)*256 + ci
__global__ void binw_kernel(const float* __restrict__ w, uint8_t* __restrict__ bout) {
  int tid = blockIdx.x * 256 + threadIdx.x;          // exactly 256*2304 threads
  int n = tid / KTOT, k = tid - n * KTOT;
  int tap = k >> 8, ci = k & 255;
  int kh = tap / 3, kw = tap - kh * 3;
  float v = w[(((size_t)n * CC + ci) * 3 + kh) * 3 + kw];
  bout[tid] = sign_fp8(v);
}

// ---- binarize acts with NCHW -> padded-NHWC transpose through LDS.
__global__ __launch_bounds__(256)
void bina_kernel(const float* __restrict__ x, uint8_t* __restrict__ apad) {
  __shared__ __align__(16) uint8_t tile[WWID * CC];  // [w][ci], 14336 B
  const int n   = blockIdx.y;
  const int hh  = blockIdx.x;                        // 0..57
  const int tid = threadIdx.x;
  uint8_t* orow = apad + ((size_t)(n * HP + hh) * HP) * CC;  // 58*256 B row

  if (hh == 0 || hh == HP - 1) {                     // fully zero border rows
    for (int i = tid; i < (HP * CC) / 16; i += 256)
      *(v4i*)(orow + (size_t)i * 16) = (v4i){0, 0, 0, 0};
    return;
  }
  const int h  = hh - 1;
  const int ci = tid;
  const float* xrow = x + (((size_t)n * CC + ci) * HH + h) * WWID;
#pragma unroll
  for (int w4 = 0; w4 < 14; ++w4) {
    float4 v = *(const float4*)(xrow + w4 * 4);
    tile[(w4 * 4 + 0) * CC + ci] = sign_fp8(v.x);
    tile[(w4 * 4 + 1) * CC + ci] = sign_fp8(v.y);
    tile[(w4 * 4 + 2) * CC + ci] = sign_fp8(v.z);
    tile[(w4 * 4 + 3) * CC + ci] = sign_fp8(v.w);
  }
  __syncthreads();
  if (tid < 32) {                                    // zero border pixels ww=0, ww=57
    int half = tid >> 4, off = (tid & 15) * 16;
    *(v4i*)(orow + (half ? (size_t)(HP - 1) * CC : 0) + off) = (v4i){0, 0, 0, 0};
  }
  // interior: 56*256 B = 896 x b128 async stores straight out of LDS
  const uint32_t lbase = (uint32_t)(uintptr_t)(&tile[0]);
#pragma unroll
  for (int j = 0; j < 3; ++j) {
    const int i = tid + j * 256;
    const uint8_t* g = orow + CC + (size_t)i * 16;
    const uint32_t l = lbase + (uint32_t)(i * 16);
    asm volatile("global_store_async_from_lds_b128 %0, %1, off"
                 :: "v"(g), "v"(l) : "memory");
  }
  if (tid < 128) {
    const int i = tid + 768;
    const uint8_t* g = orow + CC + (size_t)i * 16;
    const uint32_t l = lbase + (uint32_t)(i * 16);
    asm volatile("global_store_async_from_lds_b128 %0, %1, off"
                 :: "v"(g), "v"(l) : "memory");
  }
  asm volatile("s_wait_asynccnt 0x0" ::: "memory");
}

union Frag16 { v16i v; v4i q[4]; };

__device__ __forceinline__ void issue_btile(const uint8_t* __restrict__ bmat,
                                            int nb, int kc, uint32_t ldsdst, int tid) {
  // 64 rows x 128 B -> 512 b128 lane-ops, 2 per thread (manually unrolled)
  {
    const int rr = tid >> 3, c8 = tid & 7;
    const uint8_t* g = bmat + (size_t)(nb + rr) * KTOT + kc * 128 + c8 * 16;
    const uint32_t l = ldsdst + (uint32_t)(rr * BROW + c8 * 16);
    asm volatile("global_load_async_to_lds_b128 %0, %1, off"
                 :: "v"(l), "v"(g) : "memory");
  }
  {
    const int e = tid + 256;
    const int rr = e >> 3, c8 = e & 7;
    const uint8_t* g = bmat + (size_t)(nb + rr) * KTOT + kc * 128 + c8 * 16;
    const uint32_t l = ldsdst + (uint32_t)(rr * BROW + c8 * 16);
    asm volatile("global_load_async_to_lds_b128 %0, %1, off"
                 :: "v"(l), "v"(g) : "memory");
  }
}

__device__ __forceinline__ void load_bfrag(Frag16& bf, const uint8_t* lb,
                                           int t, int mrow, int sel) {
  const int cl = t * 16 + mrow;
#pragma unroll
  for (int q = 0; q < 4; ++q)
    bf.q[q] = *(const v4i*)(lb + cl * BROW + sel * 64 + q * 16);
}

// ---- fused implicit-GEMM conv (fp8 WMMA) + BN + BPReLU + residual + BN
// block: 256 thr (8 waves); tile 256(M) x 64(N); wave = 32M x 64N (8 WMMA / K-step)
// K loop = 18 x 128, double-buffered async LDS for B, all 4 B fragments kept live
// so ds_load_b128 groups overlap WMMA issue (staged dscnt waits).
__global__ __launch_bounds__(256)
void gemm_kernel(const uint8_t* __restrict__ apad_in,
                 const uint8_t* __restrict__ bmat,   // [n][2304] fp8 signs
                 const float* identity,
                 const float* __restrict__ bnA,      // [4][256] gamma,beta,mean,var
                 const float* __restrict__ bpA,      // [3][256] alpha,gamma,pw
                 const float* __restrict__ bnO,      // [4][256]
                 float* outbuf,
                 uint8_t* apad_out) {                // nullptr on last stage
  __shared__ __align__(16) uint8_t ldsB[2 * BBUF];  // double-buffered weight tiles

  const int tid  = threadIdx.x;
  const int lane = tid & 31;
  const int wave = tid >> 5;
  const int mrow = lane & 15;
  const int sel  = lane >> 4;
  const int nb   = blockIdx.y * 64;
  const int wbase = blockIdx.x * 256 + wave * 32;   // M total = 392*256 exactly

  // two 16-pixel A sub-tiles per wave
  const uint8_t* arow[2];
#pragma unroll
  for (int m = 0; m < 2; ++m) {
    const int p   = wbase + m * 16 + mrow;
    const int n   = p / (HH * WWID);
    const int rem = p - n * (HH * WWID);
    const int h   = rem / WWID;
    const int w   = rem - h * WWID;
    arow[m] = apad_in + ((size_t)(n * HP + h) * HP + w) * CC;
  }

  const uint32_t ldsBoff = (uint32_t)(uintptr_t)(&ldsB[0]); // low 32b = LDS offset

  v8f acc[2][4] = {};

  // prologue: start tile 0 into buffer 0
  issue_btile(bmat, nb, 0, ldsBoff, tid);

  for (int kc = 0; kc < KSTEPS; ++kc) {
    asm volatile("s_wait_asynccnt 0x0" ::: "memory");  // my buf[kc&1] copies done
    __syncthreads();                                   // all waves done filling / done reading
    if (kc + 1 < KSTEPS)                               // overlap next tile with compute
      issue_btile(bmat, nb, kc + 1, ldsBoff + (uint32_t)(((kc + 1) & 1) * BBUF), tid);

    const int tap = kc >> 1, half = kc & 1;
    const int kh = tap / 3, kw = tap - kh * 3;
    const size_t aoff = (size_t)((kh * HP + kw) * CC + half * 128 + sel * 64);
    if (kc + 1 < KSTEPS) {
      const int tap2 = (kc + 1) >> 1, half2 = (kc + 1) & 1;
      const int kh2 = tap2 / 3, kw2 = tap2 - kh2 * 3;
      const size_t aoff2 = (size_t)((kh2 * HP + kw2) * CC + half2 * 128 + sel * 64);
      __builtin_prefetch(arow[0] + aoff2, 0, 3);       // near-cache prefetch
      __builtin_prefetch(arow[1] + aoff2, 0, 3);
    }
    Frag16 af[2];
#pragma unroll
    for (int m = 0; m < 2; ++m)
#pragma unroll
      for (int q = 0; q < 4; ++q) af[m].q[q] = *(const v4i*)(arow[m] + aoff + q * 16);

    // issue all 16 ds_load_b128 first (4 live fragments) so the dscnt waits
    // before each WMMA pair are staged (<=12, <=8, <=4, 0) instead of 0 each time
    const uint8_t* lb = &ldsB[(kc & 1) * BBUF];
    Frag16 bf[4];
#pragma unroll
    for (int t = 0; t < 4; ++t) load_bfrag(bf[t], lb, t, mrow, sel);
#pragma unroll
    for (int t = 0; t < 4; ++t) {
      acc[0][t] = __builtin_amdgcn_wmma_f32_16x16x128_fp8_fp8(
          af[0].v, bf[t].v, (short)0, acc[0][t], false, false);
      acc[1][t] = __builtin_amdgcn_wmma_f32_16x16x128_fp8_fp8(
          af[1].v, bf[t].v, (short)0, acc[1][t], false, false);
    }
  }

  // ---- fused epilogue: bn -> bprelu -> +identity -> bn_out (+ re-binarize)
  float sc1[4], bb1[4], mm1[4], alv[4], gmv[4], pwv[4], sc2[4], bb2[4], mm2[4];
#pragma unroll
  for (int t = 0; t < 4; ++t) {
    const int c = nb + t * 16 + mrow;
    sc1[t] = bnA[c] * __frsqrt_rn(bnA[768 + c] + EPSV);
    bb1[t] = bnA[256 + c]; mm1[t] = bnA[512 + c];
    alv[t] = bpA[c]; gmv[t] = bpA[256 + c]; pwv[t] = bpA[512 + c];
    sc2[t] = bnO[c] * __frsqrt_rn(bnO[768 + c] + EPSV);
    bb2[t] = bnO[256 + c]; mm2[t] = bnO[512 + c];
  }
  const int cbase = nb + mrow;
#pragma unroll
  for (int m = 0; m < 2; ++m) {
#pragma unroll
    for (int r = 0; r < 8; ++r) {
      // C/D layout: VGPR r, lanes 0-15 -> M=r, lanes 16-31 -> M=r+8
      const int po = wbase + m * 16 + sel * 8 + r;
      const int nn = po / (HH * WWID);
      const int rr = po - nn * (HH * WWID);
      const int hh = rr / WWID;
      const int wo = rr - hh * WWID;
      const size_t sb = (size_t)nn * (CC * HH * WWID) + (size_t)hh * WWID + wo;
      const size_t ab = ((size_t)(nn * HP + hh + 1) * HP + (wo + 1)) * CC + cbase;
#pragma unroll
      for (int t = 0; t < 4; ++t) {
        const size_t oidx = sb + (size_t)(cbase + t * 16) * (HH * WWID);
        float y = (acc[m][t][r] - mm1[t]) * sc1[t] + bb1[t];
        y += alv[t];
        y = y >= 0.f ? y : pwv[t] * y;
        y -= gmv[t];
        y += identity[oidx];
        const float z = (y - mm2[t]) * sc2[t] + bb2[t];
        outbuf[oidx] = z;
        if (apad_out) apad_out[ab + t * 16] = sign_fp8(z);
      }
    }
  }
}

extern "C" void kernel_launch(void* const* d_in, const int* in_sizes, int n_in,
                              void* d_out, int out_size, void* d_ws, size_t ws_size,
                              hipStream_t stream) {
  const float* x    = (const float*)d_in[0];
  const float* w1   = (const float*)d_in[1];
  const float* w2   = (const float*)d_in[2];
  const float* bn1  = (const float*)d_in[3];
  const float* bno1 = (const float*)d_in[4];
  const float* bn2  = (const float*)d_in[5];
  const float* bno2 = (const float*)d_in[6];
  const float* bp1  = (const float*)d_in[7];
  const float* bp2  = (const float*)d_in[8];
  float* out = (float*)d_out;

  uint8_t* ws = (uint8_t*)d_ws;
  const size_t APAD = 32ull * HP * HP * CC;   // 27.56 MB each
  uint8_t* apad1 = ws;
  uint8_t* apad2 = ws + APAD;
  uint8_t* bm1   = ws + 2 * APAD;
  uint8_t* bm2   = bm1 + (size_t)CC * KTOT;

  // binarize weights + activations
  binw_kernel<<<KTOT, 256, 0, stream>>>(w1, bm1);
  binw_kernel<<<KTOT, 256, 0, stream>>>(w2, bm2);
  bina_kernel<<<dim3(HP, 32), 256, 0, stream>>>(x, apad1);
  hipMemsetAsync(apad2, 0, APAD, stream);     // zero borders for stage-2 pad buffer

  dim3 grid(392, 4);
  // stage 1: conv1 + bn1 + bprelu1 + x + bno1 -> mid (in d_out) + sign(mid) -> apad2
  gemm_kernel<<<grid, 256, 0, stream>>>(apad1, bm1, x, bn1, bp1, bno1, out, apad2);
  // stage 2: conv2 + bn2 + bprelu2 + mid + bno2 -> final out (in place over mid)
  gemm_kernel<<<grid, 256, 0, stream>>>(apad2, bm2, out, bn2, bp2, bno2, out, nullptr);
}
